// FeatureEncoderRelu_62947040690265
// MI455X (gfx1250) — matrix operand order repaired and checked by reading, not compiled, not run on hardware
//
#include <hip/hip_runtime.h>

#define NN      100000   // nodes
#define NE      200000   // edges
#define D_X     60
#define D_EA    50
#define D_FIN   256
#define D_FH    128
#define D_FOUT  110
#define D_FOUT_P 112     // pad 110 -> 7*16
#define D_MOD_P 224      // pad 220 -> 7*32
#define D_EMB   1200
#define X1_LD   176      // 60 + 110 padded for alignment
#define AS_LD   40       // LDS row stride (bf16 elems), 32 + 8 pad
#define MT      64       // M tile (4 waves x 16 rows)

typedef __attribute__((ext_vector_type(8)))  __bf16 bf16x8;
typedef __attribute__((ext_vector_type(16))) __bf16 bf16x16;
typedef __attribute__((ext_vector_type(8)))  float  v8f;
typedef __attribute__((ext_vector_type(4)))  unsigned int u32x4;
typedef __attribute__((ext_vector_type(8)))  int i32x8;
typedef __attribute__((ext_vector_type(4)))  int i32x4;

static __device__ inline __bf16 tobf(float f) { return (__bf16)f; }

// ---- Tensor Data Mover: 2D tile (tile_rows x 32 bf16) global -> LDS, padded to AS_LD stride ----
// D# layout per CDNA5 ISA 8.3/8.4: pad_interval code 3 = 16 dwords (32 bf16),
// pad_amount code 3 = 4 dwords (8 bf16) => LDS row stride 40 bf16.
static __device__ inline void tdm_load_2d(const void* gsrc, void* lds_dst,
                                          unsigned tile_rows, unsigned tens_rows,
                                          unsigned srcld /* bf16 elems */) {
  unsigned long long ga = (unsigned long long)(size_t)gsrc;
  unsigned la = (unsigned)(size_t)lds_dst;   // flat LDS addr: low 32 bits = LDS offset
  u32x4 g0;
  g0[0] = 1u;                                               // count=1, user descriptor
  g0[1] = la;                                               // lds_addr
  g0[2] = (unsigned)ga;                                     // global_addr[31:0]
  g0[3] = (unsigned)((ga >> 32) & 0x01FFFFFFu) | (2u << 30);// global_addr[56:32] | type=2
  i32x8 g1;
  g1[0] = (int)((1u << 16) | (1u << 20) | (3u << 22) | (3u << 25)); // data_size=2B|pad_en|intv=16dw|amt=4dw
  g1[1] = (int)((srcld & 0xFFFFu) << 16);                   // tensor_dim0[15:0]
  g1[2] = (int)((tens_rows & 0xFFFFu) << 16);               // tensor_dim0[31:16]=0 | tensor_dim1[15:0]
  g1[3] = (int)(32u << 16);                                 // tensor_dim1[31:16]=0 | tile_dim0=32
  g1[4] = (int)tile_rows;                                   // tile_dim1 | tile_dim2=0
  g1[5] = (int)srcld;                                       // tensor_dim0_stride[31:0]
  g1[6] = 0;
  g1[7] = 0;
  i32x4 z4 = {0, 0, 0, 0};
  i32x8 z8 = {0, 0, 0, 0, 0, 0, 0, 0};
  __builtin_amdgcn_tensor_load_to_lds(g0, g1, z4, z4, z8, 0);
}

// ---------------- prep kernels: fp32 -> bf16, weights TRANSPOSED (+zero pad) ----------------
__global__ void prep_weights(const float* __restrict__ W1, const float* __restrict__ W2,
                             const float* __restrict__ We,
                             __bf16* __restrict__ W1T, __bf16* __restrict__ W2T,
                             __bf16* __restrict__ WeT) {
  int i = blockIdx.x * blockDim.x + threadIdx.x;
  const int n1 = D_FH * D_FIN;        // W1T: 128 x 256
  const int n2 = D_FOUT_P * D_FH;     // W2T: 112 x 128
  const int n3 = D_EMB * D_MOD_P;     // WeT: 1200 x 224
  if (i < n1) {
    int n = i / D_FIN, k = i - n * D_FIN;
    W1T[i] = tobf(W1[k * D_FH + n]);
    return;
  }
  int j = i - n1;
  if (j < n2) {
    int n = j / D_FH, k = j - n * D_FH;
    W2T[j] = (n < D_FOUT) ? tobf(W2[k * D_FOUT + n]) : (__bf16)0.f;
    return;
  }
  int q = j - n2;
  if (q < n3) {
    int n = q / D_MOD_P, k = q - n * D_MOD_P;
    WeT[q] = (k < (D_X + D_FOUT + D_EA)) ? tobf(We[k * D_EMB + n]) : (__bf16)0.f;
  }
}

__global__ void prep_x(const float* __restrict__ x, __bf16* __restrict__ x1b) {
  int i = blockIdx.x * blockDim.x + threadIdx.x;
  if (i < NN * D_X) {
    int r = i / D_X, c = i - r * D_X;
    x1b[(long)r * X1_LD + c] = tobf(x[i]);
  }
}

__global__ void prep_ea(const float* __restrict__ ea, __bf16* __restrict__ eab) {
  int i = blockIdx.x * blockDim.x + threadIdx.x;
  if (i < NE * D_EA) eab[i] = tobf(ea[i]);
}

// ---------------- generic bf16 WMMA tile GEMM (B = transposed weights [N][K]) ----------------
// MODE 1: A = fp32 matrix (EAF), out = relu(A@B + bias) bf16
// MODE 2: A = bf16 matrix (H, staged via TDM), out = (A@B + bias) bf16 into x1[:,60:170]
// MODE 3: A = gathered rows [x1[row] | edge_attr | 0-pad], out = relu(A@B + bias) fp32
template<int MODE, int NT>
__global__ __launch_bounds__(128) void gemm_wmma(
    const float* __restrict__ Af32, const __bf16* __restrict__ Ab,
    const __bf16* __restrict__ x1, const __bf16* __restrict__ eab,
    const long long* __restrict__ erow,
    const __bf16* __restrict__ BT, const float* __restrict__ bias,
    __bf16* __restrict__ outB, float* __restrict__ outF,
    int M, int K, int out_ld, int out_off, int validN)
{
  constexpr int NTile = 16 * NT;
  __shared__ __bf16 As[MT * AS_LD];
  __shared__ __bf16 BsT[NTile * AS_LD];
  __shared__ int rows_s[MT];

  const int t    = threadIdx.x;
  const int lane = t & 31;
  const int wave = t >> 5;
  const int lr   = lane & 15;
  const int g    = lane >> 4;
  const int m0   = blockIdx.x * MT;
  const int n0   = blockIdx.y * NTile;

  if (MODE == 3) {
    if (t < MT) rows_s[t] = (int)erow[m0 + t];
  }

  v8f acc[NT];
  #pragma unroll
  for (int j = 0; j < NT; ++j)
    #pragma unroll
    for (int r = 0; r < 8; ++r) acc[j][r] = 0.f;

  for (int k0 = 0; k0 < K; k0 += 32) {
    __syncthreads();
    // ---- stage A tile [64 x 32] into LDS (bf16) ----
    if (MODE == 1) {
      #pragma unroll
      for (int it = 0; it < 2; ++it) {
        int task = t + it * 128;            // 256 tasks: 64 rows x 4 octs
        int ml = task >> 2, oct = task & 3;
        int m = m0 + ml;
        bf16x8 v;
        if (m < M) {
          const float* p = &Af32[(long)m * K + k0 + oct * 8];
          #pragma unroll
          for (int q = 0; q < 8; ++q) v[q] = tobf(p[q]);
        } else {
          #pragma unroll
          for (int q = 0; q < 8; ++q) v[q] = (__bf16)0.f;
        }
        *(bf16x8*)&As[ml * AS_LD + oct * 8] = v;
      }
    } else if (MODE == 3) {
      // gather: edge e, K layout [x1 row 0..169 | edge_attr 0..49 | pad]
      int  el  = t >> 1;
      int  kkb = (t & 1) * 16;
      int  row = rows_s[el];
      long e   = (long)m0 + el;
      #pragma unroll
      for (int half = 0; half < 2; ++half) {
        bf16x8 v;
        #pragma unroll
        for (int q = 0; q < 8; ++q) {
          int k = k0 + kkb + half * 8 + q;
          __bf16 val;
          if (k < 170)      val = x1[(long)row * X1_LD + k];
          else if (k < 220) val = eab[e * D_EA + (k - 170)];
          else              val = (__bf16)0.f;
          v[q] = val;
        }
        *(bf16x8*)&As[el * AS_LD + kkb + half * 8] = v;
      }
    }
    // ---- TDM staging (wave 0 only): B tile always, A tile for MODE 2 ----
    if (wave == 0) {
      tdm_load_2d(BT + (size_t)n0 * K + k0, (void*)BsT, NTile, NTile, (unsigned)K);
      if (MODE == 2) {
        unsigned rem = (unsigned)(M - m0);
        if (rem > MT) rem = MT;
        tdm_load_2d(Ab + (size_t)m0 * K + k0, (void*)As, MT, rem, (unsigned)K);
      }
      __builtin_amdgcn_s_wait_tensorcnt(0);
    }
    __syncthreads();
    // ---- WMMA: each wave owns one 16-row M tile x NT N tiles ----
    const __bf16* ar = &As[(wave * 16 + lr) * AS_LD + g * 8];
    bf16x8 alo = *(const bf16x8*)ar;          // K = g*8 + 0..7
    bf16x8 ahi = *(const bf16x8*)(ar + 16);   // K = g*8 + 16..23
    bf16x16 a = __builtin_shufflevector(alo, ahi,
        0,1,2,3,4,5,6,7,8,9,10,11,12,13,14,15);
    #pragma unroll
    for (int j = 0; j < NT; ++j) {
      const __bf16* br = &BsT[(j * 16 + lr) * AS_LD + g * 8];
      bf16x8 blo = *(const bf16x8*)br;        // K = g*8 + 0..7
      bf16x8 bhi = *(const bf16x8*)(br + 16); // K = g*8 + 16..23
      bf16x16 b = __builtin_shufflevector(blo, bhi,
          0,1,2,3,4,5,6,7,8,9,10,11,12,13,14,15);
      acc[j] = __builtin_amdgcn_wmma_f32_16x16x32_bf16(
          false, a, false, b, (short)0, acc[j], false, false);
    }
  }

  // ---- epilogue ----
  #pragma unroll
  for (int j = 0; j < NT; ++j) {
    int col = n0 + j * 16 + lr;
    #pragma unroll
    for (int r = 0; r < 8; ++r) {
      int row = m0 + wave * 16 + g * 8 + r;   // C VGPR r -> M = r + 8*g
      float v = acc[j][r];
      if (MODE == 1) {
        if (row < M) {
          float y = v + bias[col];
          y = y > 0.f ? y : 0.f;
          outB[(long)row * out_ld + col] = tobf(y);
        }
      } else if (MODE == 2) {
        if (row < M && col < validN) {
          outB[(long)row * out_ld + out_off + col] = tobf(v + bias[col]);
        }
      } else {
        float y = v + bias[col];
        y = y > 0.f ? y : 0.f;
        outF[(long)row * out_ld + col] = y;
      }
    }
  }
}

// ---------------- host launch ----------------
extern "C" void kernel_launch(void* const* d_in, const int* in_sizes, int n_in,
                              void* d_out, int out_size, void* d_ws, size_t ws_size,
                              hipStream_t stream) {
  (void)in_sizes; (void)n_in; (void)out_size; (void)ws_size;
  const float*     eaf  = (const float*)d_in[0];
  const float*     x    = (const float*)d_in[1];
  const long long* eidx = (const long long*)d_in[2];   // [2, NE], row = first NE
  const float*     ea   = (const float*)d_in[3];
  const float*     W1   = (const float*)d_in[4];
  const float*     b1   = (const float*)d_in[5];
  const float*     W2   = (const float*)d_in[6];
  const float*     b2   = (const float*)d_in[7];
  const float*     We   = (const float*)d_in[8];
  const float*     be   = (const float*)d_in[9];
  float* out = (float*)d_out;

  char* ws = (char*)d_ws;
  size_t off = 0;
  auto alloc = [&](size_t bytes) -> void* {
    void* p = ws + off;
    off = (off + bytes + 255) & ~(size_t)255;
    return p;
  };
  __bf16* Hb   = (__bf16*)alloc((size_t)NN * D_FH * 2);        // hidden, bf16
  __bf16* x1b  = (__bf16*)alloc((size_t)NN * X1_LD * 2);       // concat(x, enc), bf16
  __bf16* eab  = (__bf16*)alloc((size_t)NE * D_EA * 2);        // edge_attr bf16
  __bf16* W1Tb = (__bf16*)alloc((size_t)D_FH * D_FIN * 2);     // W1^T [128,256]
  __bf16* W2Tb = (__bf16*)alloc((size_t)D_FOUT_P * D_FH * 2);  // W2^T [112,128]
  __bf16* WeTb = (__bf16*)alloc((size_t)D_EMB * D_MOD_P * 2);  // We^T [1200,224]

  // prep: weight transpose/convert + activation conversion to bf16
  {
    int total = D_FH * D_FIN + D_FOUT_P * D_FH + D_EMB * D_MOD_P;
    prep_weights<<<(total + 255) / 256, 256, 0, stream>>>(W1, W2, We, W1Tb, W2Tb, WeTb);
  }
  prep_x<<<(NN * D_X + 255) / 256, 256, 0, stream>>>(x, x1b);
  prep_ea<<<(NE * D_EA + 255) / 256, 256, 0, stream>>>(ea, eab);

  // encoder layer 1: H = relu(EAF @ W1 + b1)   [100000 x 128], K=256
  gemm_wmma<1, 8><<<dim3((NN + MT - 1) / MT, 1), 128, 0, stream>>>(
      eaf, nullptr, nullptr, nullptr, nullptr, W1Tb, b1, Hb, nullptr,
      NN, D_FIN, D_FH, 0, D_FH);

  // encoder layer 2: enc = H @ W2 + b2 -> x1[:,60:170]  K=128, N=112 (valid 110)
  gemm_wmma<2, 7><<<dim3((NN + MT - 1) / MT, 1), 128, 0, stream>>>(
      nullptr, Hb, nullptr, nullptr, nullptr, W2Tb, b2, x1b, nullptr,
      NN, D_FH, X1_LD, D_X, D_FOUT);

  // edge GEMM with fused gather: h = relu(concat(x1[row], ea) @ We + be)
  // M=200000 (3125 tiles), N=1200 (25 tiles of 48), K=224
  gemm_wmma<3, 3><<<dim3(NE / MT, D_EMB / 48), 128, 0, stream>>>(
      nullptr, nullptr, x1b, eab, eidx, WeTb, be, nullptr, out,
      NE, D_MOD_P, D_EMB, 0, D_EMB);
}
